// NTXentLoss_23433341567260
// MI455X (gfx1250) — matrix operand order repaired and checked by reading, not compiled, not run on hardware
//
#include <hip/hip_runtime.h>
#include <hip/hip_bf16.h>
#include <math.h>

// ---------------------------------------------------------------------------
// NT-Xent loss for MI455X (gfx1250, wave32, WMMA f16 16x16x32).
//   z1,z2: [4096,256] f32.  N = 8192 rows after concat, D = 256.
//   sim = (zn @ zn^T) / 0.5 ; diag masked; loss = mean(-pos + LSE(sim,row)).
// Fixed-shift logsumexp (shift=2.0) is safe because sim in [-2,2].
// B tiles are staged global->LDS with async DMA (ASYNCcnt) and double-buffered
// so the DMA for tile i+1 overlaps WMMA+exp for tile i; 4 waves share each tile.
// ---------------------------------------------------------------------------

typedef _Float16 v8h  __attribute__((ext_vector_type(8)));
typedef _Float16 v16h __attribute__((ext_vector_type(16)));
typedef float    v4f  __attribute__((ext_vector_type(4)));
typedef float    v8f  __attribute__((ext_vector_type(8)));

#define BROWS 4096
#define NROWS 8192
#define DIM   256
#define TILE_BYTES 8192u   // 16 rows x 256 halves x 2B (contiguous in znh)

// ---- kernel 1: L2-normalize rows of concat(z1,z2), cast to f16 -------------
__global__ __launch_bounds__(256) void ntx_normalize(const float* __restrict__ z1,
                                                     const float* __restrict__ z2,
                                                     _Float16* __restrict__ znh) {
    const int tid  = threadIdx.x;
    const int wid  = tid >> 5;          // 8 waves -> 8 rows per block
    const int lane = tid & 31;
    const int row  = blockIdx.x * 8 + wid;
    const float* src = (row < BROWS) ? (z1 + (size_t)row * DIM)
                                     : (z2 + (size_t)(row - BROWS) * DIM);
    v4f a = *(const v4f*)(src + lane * 8);
    v4f b = *(const v4f*)(src + lane * 8 + 4);
    float ss = 0.0f;
#pragma unroll
    for (int j = 0; j < 4; ++j) ss += a[j] * a[j] + b[j] * b[j];
#pragma unroll
    for (int m = 16; m >= 1; m >>= 1) ss += __shfl_xor(ss, m, 32);
    const float inv = 1.0f / fmaxf(sqrtf(ss), 1e-8f);
    v8h o;
#pragma unroll
    for (int j = 0; j < 4; ++j) { o[j] = (_Float16)(a[j] * inv); o[4 + j] = (_Float16)(b[j] * inv); }
    *(v8h*)(znh + (size_t)row * DIM + lane * 8) = o;
}

// ---- async global->LDS staging of one contiguous 8KB B tile ----------------
// 128 threads x 4 chunks x 16B = 8192B.  GVS mode: saddr=base, vaddr=offset.
__device__ __forceinline__ void stage_tile_async(const _Float16* base,
                                                 unsigned src_byte,
                                                 unsigned lds_byte,
                                                 int tid) {
#pragma unroll
    for (int q = 0; q < 4; ++q) {
        const unsigned off = (unsigned)((q * 128 + tid) * 16);
        asm volatile("global_load_async_to_lds_b128 %0, %1, %2"
                     :: "v"(lds_byte + off), "v"(src_byte + off), "s"(base)
                     : "memory");
    }
}

// ---- kernel 2: fused sim-GEMM (WMMA) + row logsumexp + positive extraction -
// 128 threads = 4 waves; each wave owns a 16-row strip; the block shares
// double-buffered B tiles in LDS.
__global__ __launch_bounds__(128) void ntx_simlse(const _Float16* __restrict__ znh,
                                                  float* __restrict__ lse,
                                                  float* __restrict__ pos) {
    __shared__ _Float16 smem[2][16 * DIM];               // 2 x 8KB B tiles
    const int tid  = threadIdx.x;
    const int lane = tid & 31;
    const int wave = tid >> 5;
    const int m0   = (blockIdx.x * 4 + wave) * 16;       // strip base row
    const int half = lane >> 4;
    const int l15  = lane & 15;
    const unsigned lds_base = (unsigned)(size_t)(&smem[0][0]);

    // ---- preload A fragments for the whole K=256 (8 chunks of 32) ----------
    // A 16x32 f16 layout: lane holds row l15; VGPR v -> K = (v>>2)*16 + half*8 + (v&3)*2
    const _Float16* arow = znh + (size_t)(m0 + l15) * DIM;
    v16h afrag[8];
#pragma unroll
    for (int kk = 0; kk < 8; ++kk) {
        const int kc = kk * 32;
        v8h lo = *(const v8h*)(arow + kc + half * 8);
        v8h hi = *(const v8h*)(arow + kc + 16 + half * 8);
        v16h av;
#pragma unroll
        for (int j = 0; j < 8; ++j) { av[j] = lo[j]; av[8 + j] = hi[j]; }
        afrag[kk] = av;
    }

    // ---- stage tile 0 and wait -------------------------------------------
    stage_tile_async(znh, 0u, lds_base, tid);
    asm volatile("s_wait_asynccnt 0x0" ::: "memory");
    __syncthreads();

    float lsum[8];
#pragma unroll
    for (int v = 0; v < 8; ++v) lsum[v] = 0.0f;
    const int posn0 = (m0 + BROWS) & (NROWS - 1);
    float posval = 0.0f;

    for (int it = 0; it < NROWS / 16; ++it) {
        const int n0 = it * 16;
        // prefetch next tile into the other buffer (overlaps compute below)
        if (it + 1 < NROWS / 16) {
            stage_tile_async(znh, (unsigned)(n0 + 16) * 512u,
                             lds_base + (unsigned)((it + 1) & 1) * TILE_BYTES, tid);
        }

        // B 32x16 f16 = zn^T tile from LDS: lane holds column n0+l15,
        // row l15, K = kk*32 + half*8 + j (j<8) and kk*32 + 16 + half*8 + (j-8).
        const _Float16* btile = &smem[it & 1][0];
        v8f c = {};
#pragma unroll
        for (int kk = 0; kk < 8; ++kk) {
            v8h blo = *(const v8h*)(btile + l15 * DIM + kk * 32 + half * 8);
            v8h bhi = *(const v8h*)(btile + l15 * DIM + kk * 32 + 16 + half * 8);
            v16h b;
#pragma unroll
            for (int j = 0; j < 8; ++j) { b[j] = blo[j]; b[8 + j] = bhi[j]; }
            c = __builtin_amdgcn_wmma_f32_16x16x32_f16(false, afrag[kk], false, b,
                                                       (short)0, c, false, false);
        }
        const bool isSelf = (n0 == m0);
        const bool isPos  = (n0 == posn0);
#pragma unroll
        for (int v = 0; v < 8; ++v) {
            const float sim  = 2.0f * c[v];              // 1/temperature
            const bool  diag = (l15 == v + 8 * half);    // C layout: row v+8*half
            if (isPos && diag) posval = sim;
            const float e = __expf(sim - 2.0f);          // shift=2: args in [-4,0]
            lsum[v] += (isSelf && diag) ? 0.0f : e;      // masked diag adds 0
        }

        asm volatile("s_wait_asynccnt 0x0" ::: "memory");
        __syncthreads();
    }

    // ---- reduce sums across the 16 lanes holding each row ------------------
#pragma unroll
    for (int v = 0; v < 8; ++v) {
        float s = lsum[v];
        s += __shfl_xor(s, 1, 32);
        s += __shfl_xor(s, 2, 32);
        s += __shfl_xor(s, 4, 32);
        s += __shfl_xor(s, 8, 32);
        lsum[v] = s;
    }
    if (l15 == 0) {                                      // lanes 0 and 16
#pragma unroll
        for (int v = 0; v < 8; ++v)
            lse[m0 + v + 8 * half] = 2.0f + __logf(lsum[v]);
    }
    // diagonal holders: half0 -> lanes 0..7 (row l15), half1 -> lanes 24..31
    if ((half == 0 && l15 < 8) || (half == 1 && l15 >= 8))
        pos[m0 + l15] = posval;
}

// ---- kernel 3: deterministic final reduction -> mean loss ------------------
__global__ __launch_bounds__(256) void ntx_reduce(const float* __restrict__ lse,
                                                  const float* __restrict__ pos,
                                                  float* __restrict__ out) {
    __shared__ float sm[256];
    float acc = 0.0f;
    for (int i = threadIdx.x; i < NROWS; i += 256) acc += lse[i] - pos[i];
    sm[threadIdx.x] = acc;
    __syncthreads();
    for (int s = 128; s >= 1; s >>= 1) {
        if (threadIdx.x < s) sm[threadIdx.x] += sm[threadIdx.x + s];
        __syncthreads();
    }
    if (threadIdx.x == 0) out[0] = sm[0] * (1.0f / (float)NROWS);
}

extern "C" void kernel_launch(void* const* d_in, const int* in_sizes, int n_in,
                              void* d_out, int out_size, void* d_ws, size_t ws_size,
                              hipStream_t stream) {
    (void)in_sizes; (void)n_in; (void)out_size; (void)ws_size;
    const float* z1 = (const float*)d_in[0];
    const float* z2 = (const float*)d_in[1];

    // workspace layout: [f16 zn 8192x256 = 4MB][f32 lse 8192][f32 pos 8192]
    _Float16* znh = (_Float16*)d_ws;
    float*    lse = (float*)((char*)d_ws + (size_t)NROWS * DIM * sizeof(_Float16));
    float*    pos = lse + NROWS;

    ntx_normalize<<<NROWS / 8, 256, 0, stream>>>(z1, z2, znh);
    ntx_simlse<<<NROWS / 64, 128, 0, stream>>>(znh, lse, pos);
    ntx_reduce<<<1, 256, 0, stream>>>(lse, pos, (float*)d_out);
}
